// Head_80504866996370
// MI455X (gfx1250) — compile-verified, matrix operand order checked by reading, not run on hardware
//
#include <hip/hip_runtime.h>

typedef __attribute__((ext_vector_type(16))) _Float16 v16h;
typedef __attribute__((ext_vector_type(8)))  _Float16 v8h;
typedef __attribute__((ext_vector_type(8)))  float    v8f;

static constexpr int Bn = 4;
static constexpr int Tn = 4096;
static constexpr int Cn = 1024;
static constexpr int HS = 128;

// LDS tile strides (halves), padded for bank-conflict-free A-fragment reads.
static constexpr int KSTR = 136;   // 128 + 8  -> row stride 272B (4*r mod 64 banks)
static constexpr int VSTR = 40;    // 32 + 8   -> row stride 80B  (20*r mod 64 banks)

static __device__ __forceinline__ v16h cat8(v8h a, v8h b) {
  return __builtin_shufflevector(a, b, 0,1,2,3,4,5,6,7,8,9,10,11,12,13,14,15);
}

static __device__ __forceinline__ v8f wmma_f16(v16h a, v16h b, v8f c) {
  return __builtin_amdgcn_wmma_f32_16x16x32_f16(false, a, false, b, (short)0, c,
                                                false, false);
}

// ---------------------------------------------------------------------------
// Kernel 0: transpose + convert weights.  W[C][HS] f32 -> Wt[HS][C] f16.
// The attention score scale C^-0.5 = 1/32 is folded into Wt_q.
// ---------------------------------------------------------------------------
__global__ __launch_bounds__(256) void wconv_kernel(
    const float* __restrict__ Wq, const float* __restrict__ Wk,
    const float* __restrict__ Wv, _Float16* __restrict__ Wt)
{
  const int idx = blockIdx.x * 256 + threadIdx.x;   // 0 .. 3*Cn*HS-1
  const int w   = idx / (Cn * HS);
  const int r   = idx % (Cn * HS);
  const int h   = r / Cn;
  const int c   = r % Cn;
  const float* src = (w == 0) ? Wq : (w == 1) ? Wk : Wv;
  const float scale = (w == 0) ? 0.03125f : 1.0f;
  Wt[idx] = (_Float16)(src[(size_t)c * HS + h] * scale);
}

// ---------------------------------------------------------------------------
// Kernel 1: fused QKV projection with WMMA 16x16x32 f16 / f32 acc.
// ---------------------------------------------------------------------------
__global__ __launch_bounds__(32) void proj_qkv_kernel(
    const float* __restrict__ x, const _Float16* __restrict__ Wt,
    _Float16* __restrict__ Qh, _Float16* __restrict__ Kh,
    _Float16* __restrict__ Vt)
{
  const int lane  = threadIdx.x;
  const int mrow  = lane & 15;
  const int hi    = lane >> 4;
  const int mtile = blockIdx.x;
  const int ntile = blockIdx.y;
  const int b     = mtile / (Tn / 16);
  const int tbase = (mtile % (Tn / 16)) * 16;
  const int hbase = ntile * 16;

  const float* xrow = x + ((size_t)(b * Tn + tbase + mrow)) * Cn;
  const _Float16* wq = Wt + (size_t)(hbase + mrow) * Cn;
  const _Float16* wk = wq + (size_t)HS * Cn;
  const _Float16* wv = wk + (size_t)HS * Cn;

  v8f cq = {}; v8f ck = {}; v8f cv = {};

  for (int kb = 0; kb < Cn; kb += 32) {
    const int c0 = kb + hi * 8;
    float4 r0 = *(const float4*)(xrow + c0);
    float4 r1 = *(const float4*)(xrow + c0 + 4);
    float4 r2 = *(const float4*)(xrow + c0 + 16);
    float4 r3 = *(const float4*)(xrow + c0 + 20);
    v16h a;
    a[0]=(_Float16)r0.x; a[1]=(_Float16)r0.y; a[2]=(_Float16)r0.z; a[3]=(_Float16)r0.w;
    a[4]=(_Float16)r1.x; a[5]=(_Float16)r1.y; a[6]=(_Float16)r1.z; a[7]=(_Float16)r1.w;
    a[8]=(_Float16)r2.x; a[9]=(_Float16)r2.y; a[10]=(_Float16)r2.z; a[11]=(_Float16)r2.w;
    a[12]=(_Float16)r3.x; a[13]=(_Float16)r3.y; a[14]=(_Float16)r3.z; a[15]=(_Float16)r3.w;

    v16h bq = cat8(*(const v8h*)(wq + c0), *(const v8h*)(wq + c0 + 16));
    v16h bk = cat8(*(const v8h*)(wk + c0), *(const v8h*)(wk + c0 + 16));
    v16h bv = cat8(*(const v8h*)(wv + c0), *(const v8h*)(wv + c0 + 16));

    cq = wmma_f16(a, bq, cq);
    ck = wmma_f16(a, bk, ck);
    cv = wmma_f16(a, bv, cv);
  }

  const int h = hbase + mrow;
#pragma unroll
  for (int r = 0; r < 8; ++r) {
    const int tt = tbase + r + hi * 8;
    const size_t ridx = ((size_t)(b * Tn + tt)) * HS + h;
    Qh[ridx] = (_Float16)cq[r];
    Kh[ridx] = (_Float16)ck[r];
  }
  v8h vpack;
#pragma unroll
  for (int r = 0; r < 8; ++r) vpack[r] = (_Float16)cv[r];
  *(v8h*)(Vt + ((size_t)(b * HS + h)) * Tn + tbase + hi * 8) = vpack;
}

// ---------------------------------------------------------------------------
// Kernel 2: causal flash attention, one wave per 16-query tile.
// S^T = K*Q^T (lane-local softmax); exp(S^T) C-layout == P B-layout.
// K/V tiles double-buffered in LDS via GLOBAL_LOAD_ASYNC_TO_LDS_B128 so the
// next tile's DMA overlaps the current tile's WMMAs + softmax.
// ---------------------------------------------------------------------------
__global__ __launch_bounds__(32) void attn_kernel(
    const _Float16* __restrict__ Qh, const _Float16* __restrict__ Kh,
    const _Float16* __restrict__ Vt, float* __restrict__ out)
{
  __shared__ _Float16 Kbuf[2][32][KSTR];    // 17,408 B
  __shared__ _Float16 Vbuf[2][HS][VSTR];    // 20,480 B

  const int lane  = threadIdx.x;
  const int mrow  = lane & 15;
  const int hi    = lane >> 4;
  const int qtile = gridDim.x - 1 - blockIdx.x;   // longest tiles first
  const int b     = blockIdx.y;
  const int qbase = qtile * 16;

  // ---- per-lane async-copy addressing (lane-dependent components) --------
  // K tile copy: instr i moves rows 2i..2i+1; lane -> row=(lane>>4), 16B chunk=lane&15
  const uint64_t kg0 =
      (uint64_t)(uintptr_t)(Kh + ((size_t)(b * Tn + (lane >> 4))) * HS + (lane & 15) * 8);
  uint32_t klds0[2];
  klds0[0] = (uint32_t)(uintptr_t)&Kbuf[0][lane >> 4][(lane & 15) * 8];
  klds0[1] = (uint32_t)(uintptr_t)&Kbuf[1][lane >> 4][(lane & 15) * 8];
  // V tile copy: instr i moves rows 8i..8i+7; lane -> row=(lane>>2), chunk=lane&3
  const uint64_t vg0 =
      (uint64_t)(uintptr_t)(Vt + ((size_t)(b * HS + (lane >> 2))) * Tn + (lane & 3) * 8);
  uint32_t vlds0[2];
  vlds0[0] = (uint32_t)(uintptr_t)&Vbuf[0][lane >> 2][(lane & 3) * 8];
  vlds0[1] = (uint32_t)(uintptr_t)&Vbuf[1][lane >> 2][(lane & 3) * 8];

  // Issue async DMA of the 32-key K and V tiles starting at key kb into buf p.
  auto issue_tile = [&](int kb, int p) {
    const uint64_t kg = kg0 + (uint64_t)kb * (HS * 2);
#pragma unroll
    for (int i = 0; i < 16; ++i) {
      // asm offset adds to BOTH global and LDS addr; LDS operand carries only
      // the padding delta (2 rows: 544B lds vs 512B global -> +32B per instr).
      const uint32_t l = klds0[p] + (uint32_t)(i * 32);
      asm volatile("global_load_async_to_lds_b128 %0, %1, off offset:%2"
                   :: "v"(l), "v"(kg), "i"(i * 512) : "memory");
    }
#pragma unroll
    for (int i = 0; i < 16; ++i) {
      const uint64_t g = vg0 + (uint64_t)kb * 2 + (uint64_t)i * 8 * Tn * 2;
      const uint32_t l = vlds0[p] + (uint32_t)(i * 8 * VSTR * 2);
      asm volatile("global_load_async_to_lds_b128 %0, %1, off"
                   :: "v"(l), "v"(g) : "memory");
    }
  };

  // ---- Q^T B-fragments (loop invariant, direct global loads) -------------
  const _Float16* qrow = Qh + ((size_t)(b * Tn + qbase + mrow)) * HS;
  v16h qf[4];
#pragma unroll
  for (int c = 0; c < 4; ++c) {
    const int h0 = c * 32 + hi * 8;
    qf[c] = cat8(*(const v8h*)(qrow + h0), *(const v8h*)(qrow + h0 + 16));
  }

  v8f o[8];
#pragma unroll
  for (int t = 0; t < 8; ++t) o[t] = (v8f){};
  float mrun = -1e30f, lrun = 0.0f;

  const int kend = qbase + 16;               // causal bound (exclusive)
  issue_tile(0, 0);                          // prologue DMA

  for (int kb = 0; kb < kend; kb += 32) {
    const int p = (kb >> 5) & 1;
    asm volatile("s_wait_asynccnt 0x0" ::: "memory");

    // ---- scores S^T for 32 keys from LDS K tile
    v8f s0 = {}; v8f s1 = {};
#pragma unroll
    for (int c = 0; c < 4; ++c) {
      const int hoff = c * 32 + hi * 8;
      v16h ka0 = cat8(*(const v8h*)&Kbuf[p][mrow][hoff],
                      *(const v8h*)&Kbuf[p][mrow][hoff + 16]);
      s0 = wmma_f16(ka0, qf[c], s0);
      v16h ka1 = cat8(*(const v8h*)&Kbuf[p][mrow + 16][hoff],
                      *(const v8h*)&Kbuf[p][mrow + 16][hoff + 16]);
      s1 = wmma_f16(ka1, qf[c], s1);
    }

    // ---- kick off next tile's DMA; overlaps softmax + PV below
    if (kb + 32 < kend) issue_tile(kb + 32, p ^ 1);

    // ---- causal mask (diagonal steps only; uniform branch)
    const int qg = qbase + mrow;
    if (kb + 31 > qbase) {
#pragma unroll
      for (int r = 0; r < 8; ++r) {
        if (kb + r + hi * 8 > qg)      s0[r] = -1e30f;
        if (kb + 16 + r + hi * 8 > qg) s1[r] = -1e30f;
      }
    }

    // ---- online softmax (lane-local + one xor-16 merge)
    float mx = s0[0];
#pragma unroll
    for (int r = 1; r < 8; ++r) mx = fmaxf(mx, s0[r]);
#pragma unroll
    for (int r = 0; r < 8; ++r) mx = fmaxf(mx, s1[r]);
    mx = fmaxf(mx, __shfl_xor(mx, 16, 32));
    const float mnew  = fmaxf(mrun, mx);
    const float alpha = __expf(mrun - mnew);

    float rsum = 0.0f;
    v16h pf;
#pragma unroll
    for (int r = 0; r < 8; ++r) {
      const float p0 = __expf(s0[r] - mnew);
      const float p1 = __expf(s1[r] - mnew);
      rsum += p0 + p1;
      pf[r]     = (_Float16)p0;
      pf[r + 8] = (_Float16)p1;
    }
    rsum += __shfl_xor(rsum, 16, 32);
    lrun = lrun * alpha + rsum;
    mrun = mnew;

#pragma unroll
    for (int t = 0; t < 8; ++t)
#pragma unroll
      for (int r = 0; r < 8; ++r) o[t][r] *= alpha;

    // ---- O^T += V^T * P from LDS V tile
#pragma unroll
    for (int t = 0; t < 8; ++t) {
      const int vr = t * 16 + mrow;
      v16h va = cat8(*(const v8h*)&Vbuf[p][vr][hi * 8],
                     *(const v8h*)&Vbuf[p][vr][hi * 8 + 16]);
      o[t] = wmma_f16(va, pf, o[t]);
    }
  }

  // ---- epilogue: out[b][q][h] = O^T[h][q] / l(q); h contiguous in r
  const float inv = 1.0f / lrun;
  float* orow = out + ((size_t)(b * Tn + qbase + mrow)) * HS;
#pragma unroll
  for (int t = 0; t < 8; ++t) {
    float4 f0, f1;
    f0.x = o[t][0] * inv; f0.y = o[t][1] * inv;
    f0.z = o[t][2] * inv; f0.w = o[t][3] * inv;
    f1.x = o[t][4] * inv; f1.y = o[t][5] * inv;
    f1.z = o[t][6] * inv; f1.w = o[t][7] * inv;
    *(float4*)(orow + t * 16 + hi * 8)     = f0;
    *(float4*)(orow + t * 16 + hi * 8 + 4) = f1;
  }
}

extern "C" void kernel_launch(void* const* d_in, const int* in_sizes, int n_in,
                              void* d_out, int out_size, void* d_ws, size_t ws_size,
                              hipStream_t stream) {
  const float* x  = (const float*)d_in[0];
  const float* Wq = (const float*)d_in[1];
  const float* Wk = (const float*)d_in[2];
  const float* Wv = (const float*)d_in[3];
  float* out = (float*)d_out;

  const size_t nqkv = (size_t)Bn * Tn * HS;
  _Float16* qh = (_Float16*)d_ws;
  _Float16* kh = qh + nqkv;
  _Float16* vt = kh + nqkv;
  _Float16* wt = vt + nqkv;                   // 3 * HS * Cn f16

  const int nw = 3 * Cn * HS;
  wconv_kernel<<<nw / 256, 256, 0, stream>>>(Wq, Wk, Wv, wt);

  dim3 gridP(Bn * Tn / 16, HS / 16);
  proj_qkv_kernel<<<gridP, 32, 0, stream>>>(x, wt, qh, kh, vt);

  dim3 gridA(Tn / 16, Bn);
  attn_kernel<<<gridA, 32, 0, stream>>>(qh, kh, vt, out);
}
